// GATLayer_45638322487926
// MI455X (gfx1250) — matrix operand order repaired and checked by reading, not compiled, not run on hardware
//
#include <hip/hip_runtime.h>
#include <hip/hip_bf16.h>

// ---------------------------------------------------------------------------
// GAT layer for MI455X (gfx1250, wave32).
//   h = x@W via v_wmma_f32_16x16x32_f16 (fp16 in / fp32 acc),
//     W staged in LDS pre-swizzled into the WMMA B-fragment layout so each
//     fragment is a single 32B contiguous LDS read (2x ds_load_b128).
//   edge softmax + scatter-add via L2-resident f32 atomics
//   LayerNorm/PReLU via wave32 shuffle reductions
// ---------------------------------------------------------------------------

typedef __attribute__((ext_vector_type(16))) _Float16 v16h;
typedef __attribute__((ext_vector_type(8)))  _Float16 v8h;
typedef __attribute__((ext_vector_type(8)))  float    v8f;

#define IN_DIM   128
#define OUT_DIM  128
#define HEADS    4
#define NEG_SLOPE 0.2f
#define LN_EPS   1e-5f

__device__ __forceinline__ float leaky(float v) {
    return v >= 0.0f ? v : NEG_SLOPE * v;
}

// ---------------------------------------------------------------------------
// Kernel 1: h = x @ W  (WMMA, fp16 staged in LDS, fp32 accumulate)
// Block = 128 threads = 4 waves; each wave owns a 16-row strip, full 128 cols.
// ---------------------------------------------------------------------------
#define XPAD 136   // x-tile LDS row pitch in halfs (16B-aligned rows, no conflicts)

__global__ __launch_bounds__(128)
void gemm_h_kernel(const float* __restrict__ x, const float* __restrict__ W,
                   float* __restrict__ hbuf, int n)
{
    __shared__ _Float16 sx[64 * XPAD];       // x tile, row-major + pad (17408 B)
    // W pre-swizzled into B-fragment layout: [ct][kc][lane][16 halfs] (32768 B)
    __shared__ _Float16 swf[8 * 4 * 32 * 16];

    const int tid  = threadIdx.x;
    const int wave = tid >> 5;
    const int lane = tid & 31;
    const int rowBase = blockIdx.x * 64;

    // ---- Stage W in WMMA B-fragment order; thread t owns output column t.
    // B layout matches A: fragment(ct,kc) lane (hf*16+ml) holds K
    // [kc*32+hf*8, +8) then [kc*32+16+hf*8, +8) of column ct*16+ml.
    {
        const int col = tid;                 // 0..127
        const int ct  = col >> 4;
        const int ml  = col & 15;
#pragma unroll
        for (int kc = 0; kc < 4; ++kc) {
#pragma unroll
            for (int hf = 0; hf < 2; ++hf) {
                const int kBase = kc * 32 + hf * 8;
                v16h tmp;
#pragma unroll
                for (int j = 0; j < 16; ++j)      // coalesced across lanes
                    tmp[j] = (_Float16)W[(kBase + j + (j >= 8 ? 8 : 0)) * OUT_DIM + col];
                *(v16h*)&swf[(((ct * 4 + kc) * 32) + hf * 16 + ml) * 16] = tmp;
            }
        }
    }
    // ---- Stage x tile as fp16 (zero-pad rows past n)
    for (int i = tid; i < 64 * 128; i += 128) {
        int r = i >> 7, c = i & 127;
        int gr = rowBase + r;
        sx[r * XPAD + c] = (gr < n) ? (_Float16)x[(long)gr * IN_DIM + c]
                                    : (_Float16)0.0f;
    }
    __syncthreads();

    const int halfSel = lane >> 4;   // 0: lanes 0-15, 1: lanes 16-31
    const int ml      = lane & 15;
    const int aRow    = wave * 16 + ml;
    const bool full   = (rowBase + 64 <= n);       // block-uniform tail check

    // ---- A fragments (ISA 16-bit A 16x32 layout): per (lane, kc) the frag is
    // two contiguous 8-half runs: K [kc*32+hf*8, +8) and K [kc*32+16+hf*8, +8).
    v16h afr[4];
#pragma unroll
    for (int kc = 0; kc < 4; ++kc) {
        const _Float16* rp = &sx[aRow * XPAD + kc * 32 + halfSel * 8];
        v8h c0 = *(const v8h*)(rp);
        v8h c1 = *(const v8h*)(rp + 16);
        afr[kc] = __builtin_shufflevector(c0, c1, 0, 1, 2, 3, 4, 5, 6, 7,
                                          8, 9, 10, 11, 12, 13, 14, 15);
    }

    // Per-lane base row for C/D stores: M = r + 8*halfSel, N = ml
    const int rowLane = rowBase + wave * 16 + 8 * halfSel;

#pragma unroll
    for (int ct = 0; ct < 8; ++ct) {           // 8 column tiles of 16
        v8f acc = {};
#pragma unroll
        for (int kc = 0; kc < 4; ++kc) {
            // One aligned 32B contiguous LDS read -> 2x ds_load_b128
            v16h bfr = *(const v16h*)&swf[(((ct * 4 + kc) * 32) + lane) * 16];
            acc = __builtin_amdgcn_wmma_f32_16x16x32_f16(
                false, afr[kc], false, bfr, (short)0, acc, false, false);
        }
        // Stores: constant 512B row stride from one base -> immediate offsets
        float* op = hbuf + (long)rowLane * OUT_DIM + ct * 16 + ml;
        if (full) {
#pragma unroll
            for (int r = 0; r < 8; ++r) op[r * OUT_DIM] = acc[r];
        } else {
#pragma unroll
            for (int r = 0; r < 8; ++r)
                if (rowLane + r < n) op[r * OUT_DIM] = acc[r];
        }
    }
}

// ---------------------------------------------------------------------------
// Kernel 2: per-node attention halves a_s/a_d (wave per node; 8 lanes/head)
// ---------------------------------------------------------------------------
__global__ __launch_bounds__(256)
void att_kernel(const float* __restrict__ hbuf, const float* __restrict__ att_src,
                const float* __restrict__ att_dst, float* __restrict__ a_s,
                float* __restrict__ a_d, int n)
{
    int node = (int)((blockIdx.x * (unsigned)blockDim.x + threadIdx.x) >> 5);
    int lane = threadIdx.x & 31;
    if (node >= n) return;
    float4 hv  = ((const float4*)(hbuf + (long)node * OUT_DIM))[lane];
    float4 asv = ((const float4*)att_src)[lane];
    float4 adv = ((const float4*)att_dst)[lane];
    float ps = hv.x * asv.x + hv.y * asv.y + hv.z * asv.z + hv.w * asv.w;
    float pd = hv.x * adv.x + hv.y * adv.y + hv.z * adv.z + hv.w * adv.w;
#pragma unroll
    for (int off = 1; off < 8; off <<= 1) {     // reduce within 8-lane head group
        ps += __shfl_xor(ps, off, 32);
        pd += __shfl_xor(pd, off, 32);
    }
    if ((lane & 7) == 0) {
        int head = lane >> 3;
        a_s[node * HEADS + head] = ps;
        a_d[node * HEADS + head] = pd;
    }
}

// ---------------------------------------------------------------------------
// Kernel 3: zero the output accumulator (d_out doubles as segment-sum buffer)
// ---------------------------------------------------------------------------
__global__ void zero_kernel(float4* __restrict__ p, int n4)
{
    int i = blockIdx.x * blockDim.x + threadIdx.x;
    if (i < n4) p[i] = make_float4(0.f, 0.f, 0.f, 0.f);
}

// ---------------------------------------------------------------------------
// Kernel 4: init segment-max with the self-loop score (exactly one per node)
// ---------------------------------------------------------------------------
__global__ void node_max_init_kernel(const float* __restrict__ a_s,
                                     const float* __restrict__ a_d,
                                     float* __restrict__ m, int n)
{
    int i = blockIdx.x * blockDim.x + threadIdx.x;
    if (i >= n) return;
    float4 s = ((const float4*)a_s)[i];
    float4 d = ((const float4*)a_d)[i];
    float4 e = make_float4(leaky(s.x + d.x), leaky(s.y + d.y),
                           leaky(s.z + d.z), leaky(s.w + d.w));
    ((float4*)m)[i] = e;
}

// ---------------------------------------------------------------------------
// Kernel 5: per-edge segment max (float atomicMax, L2-resident)
// ---------------------------------------------------------------------------
__global__ void edge_max_kernel(const int* __restrict__ src, const int* __restrict__ dst,
                                const float* __restrict__ a_s, const float* __restrict__ a_d,
                                float* __restrict__ m, int e)
{
    int t = blockIdx.x * blockDim.x + threadIdx.x;
    if (t >= e) return;
    int s = src[t], d = dst[t];
    float4 sv = ((const float4*)a_s)[s];
    float4 dv = ((const float4*)a_d)[d];
    float* md = m + (long)d * HEADS;
    atomicMax(md + 0, leaky(sv.x + dv.x));
    atomicMax(md + 1, leaky(sv.y + dv.y));
    atomicMax(md + 2, leaky(sv.z + dv.z));
    atomicMax(md + 3, leaky(sv.w + dv.w));
}

// ---------------------------------------------------------------------------
// Kernel 6: seed denom with the self-loop probability (after max is final)
// ---------------------------------------------------------------------------
__global__ void node_denom_init_kernel(const float* __restrict__ a_s,
                                       const float* __restrict__ a_d,
                                       const float* __restrict__ m,
                                       float* __restrict__ denom, int n)
{
    int i = blockIdx.x * blockDim.x + threadIdx.x;
    if (i >= n) return;
    float4 s  = ((const float4*)a_s)[i];
    float4 d  = ((const float4*)a_d)[i];
    float4 mv = ((const float4*)m)[i];
    float4 p;
    p.x = __expf(leaky(s.x + d.x) - mv.x);
    p.y = __expf(leaky(s.y + d.y) - mv.y);
    p.z = __expf(leaky(s.z + d.z) - mv.z);
    p.w = __expf(leaky(s.w + d.w) - mv.w);
    ((float4*)denom)[i] = p;
}

// ---------------------------------------------------------------------------
// Kernel 7: per-edge p = exp(e - m[dst]); atomicAdd into denom; stash p
// ---------------------------------------------------------------------------
__global__ void edge_exp_kernel(const int* __restrict__ src, const int* __restrict__ dst,
                                const float* __restrict__ a_s, const float* __restrict__ a_d,
                                const float* __restrict__ m, float* __restrict__ denom,
                                float* __restrict__ pbuf, int e)
{
    int t = blockIdx.x * blockDim.x + threadIdx.x;
    if (t >= e) return;
    int s = src[t], d = dst[t];
    float4 sv = ((const float4*)a_s)[s];
    float4 dv = ((const float4*)a_d)[d];
    float4 mv = ((const float4*)m)[d];
    float4 p;
    p.x = __expf(leaky(sv.x + dv.x) - mv.x);
    p.y = __expf(leaky(sv.y + dv.y) - mv.y);
    p.z = __expf(leaky(sv.z + dv.z) - mv.z);
    p.w = __expf(leaky(sv.w + dv.w) - mv.w);
    ((float4*)pbuf)[t] = p;
    float* dd = denom + (long)d * HEADS;
    atomicAdd(dd + 0, p.x);
    atomicAdd(dd + 1, p.y);
    atomicAdd(dd + 2, p.z);
    atomicAdd(dd + 3, p.w);
}

// ---------------------------------------------------------------------------
// Kernel 8: weighted scatter-add, one wave32 per edge (512B row per edge).
// Lane l handles features [4l, 4l+4), head = l/8. 4 f32 atomics per lane.
// ---------------------------------------------------------------------------
__global__ __launch_bounds__(256)
void edge_agg_kernel(const int* __restrict__ src, const int* __restrict__ dst,
                     const float* __restrict__ hbuf, const float* __restrict__ pbuf,
                     const float* __restrict__ denom, float* __restrict__ out, int e)
{
    int edge = (int)((blockIdx.x * (unsigned)blockDim.x + threadIdx.x) >> 5);
    int lane = threadIdx.x & 31;
    if (edge >= e) return;
    int s = src[edge], d = dst[edge];
    int head = lane >> 3;
    float alpha = pbuf[(long)edge * HEADS + head] /
                  (denom[(long)d * HEADS + head] + 1e-16f);
    float4 hv = ((const float4*)(hbuf + (long)s * OUT_DIM))[lane];
    float* o = out + (long)d * OUT_DIM + lane * 4;
    atomicAdd(o + 0, hv.x * alpha);
    atomicAdd(o + 1, hv.y * alpha);
    atomicAdd(o + 2, hv.z * alpha);
    atomicAdd(o + 3, hv.w * alpha);
}

// ---------------------------------------------------------------------------
// Kernel 9: add self-loop contribution + bias, LayerNorm (wave32 reduce), PReLU
// ---------------------------------------------------------------------------
__global__ __launch_bounds__(256)
void final_kernel(const float* __restrict__ hbuf, const float* __restrict__ a_s,
                  const float* __restrict__ a_d, const float* __restrict__ m,
                  const float* __restrict__ denom, const float* __restrict__ bias,
                  const float* __restrict__ ln_w, const float* __restrict__ ln_b,
                  const float* __restrict__ prelu_w, float* __restrict__ out, int n)
{
    int node = (int)((blockIdx.x * (unsigned)blockDim.x + threadIdx.x) >> 5);
    int lane = threadIdx.x & 31;
    if (node >= n) return;
    int head = lane >> 3;
    float el = leaky(a_s[node * HEADS + head] + a_d[node * HEADS + head]);
    float dn = denom[(long)node * HEADS + head] + 1e-16f;
    float alpha = __expf(el - m[(long)node * HEADS + head]) / dn;

    float4 acc = ((const float4*)(out  + (long)node * OUT_DIM))[lane];
    float4 hv  = ((const float4*)(hbuf + (long)node * OUT_DIM))[lane];
    float4 bv  = ((const float4*)bias)[lane];
    float v0 = acc.x + hv.x * alpha + bv.x;
    float v1 = acc.y + hv.y * alpha + bv.y;
    float v2 = acc.z + hv.z * alpha + bv.z;
    float v3 = acc.w + hv.w * alpha + bv.w;

    float sum = v0 + v1 + v2 + v3;
    float sq  = v0 * v0 + v1 * v1 + v2 * v2 + v3 * v3;
#pragma unroll
    for (int off = 1; off < 32; off <<= 1) {
        sum += __shfl_xor(sum, off, 32);
        sq  += __shfl_xor(sq,  off, 32);
    }
    float mu  = sum * (1.0f / OUT_DIM);
    float var = sq * (1.0f / OUT_DIM) - mu * mu;
    float inv = rsqrtf(var + LN_EPS);
    float slope = prelu_w[0];

    float4 w4 = ((const float4*)ln_w)[lane];
    float4 b4 = ((const float4*)ln_b)[lane];
    float4 y;
    y.x = (v0 - mu) * inv * w4.x + b4.x;  y.x = y.x >= 0.f ? y.x : slope * y.x;
    y.y = (v1 - mu) * inv * w4.y + b4.y;  y.y = y.y >= 0.f ? y.y : slope * y.y;
    y.z = (v2 - mu) * inv * w4.z + b4.z;  y.z = y.z >= 0.f ? y.z : slope * y.z;
    y.w = (v3 - mu) * inv * w4.w + b4.w;  y.w = y.w >= 0.f ? y.w : slope * y.w;
    ((float4*)(out + (long)node * OUT_DIM))[lane] = y;
}

// ---------------------------------------------------------------------------
extern "C" void kernel_launch(void* const* d_in, const int* in_sizes, int n_in,
                              void* d_out, int out_size, void* d_ws, size_t ws_size,
                              hipStream_t stream)
{
    const float* x       = (const float*)d_in[0];
    const int*   edge    = (const int*)  d_in[1];
    const float* W       = (const float*)d_in[2];
    const float* att_src = (const float*)d_in[3];
    const float* att_dst = (const float*)d_in[4];
    const float* bias    = (const float*)d_in[5];
    const float* ln_w    = (const float*)d_in[6];
    const float* ln_b    = (const float*)d_in[7];
    const float* prelu_w = (const float*)d_in[8];

    const int n = in_sizes[0] / IN_DIM;     // 50000
    const int e = in_sizes[1] / 2;          // 800000
    const int* src = edge;                  // edge_index[0]
    const int* dst = edge + e;              // edge_index[1]
    float* out = (float*)d_out;

    // Workspace carve-up (256B aligned): h, a_s, a_d, m, denom, p  (~42 MB)
    char* ws = (char*)d_ws;
    size_t off = 0;
    auto carve = [&](size_t bytes) -> float* {
        float* p = (float*)(ws + off);
        off = (off + bytes + 255) & ~(size_t)255;
        return p;
    };
    float* hbuf  = carve((size_t)n * OUT_DIM * sizeof(float));
    float* a_s   = carve((size_t)n * HEADS * sizeof(float));
    float* a_d   = carve((size_t)n * HEADS * sizeof(float));
    float* mbuf  = carve((size_t)n * HEADS * sizeof(float));
    float* denom = carve((size_t)n * HEADS * sizeof(float));
    float* pbuf  = carve((size_t)e * HEADS * sizeof(float));
    (void)ws_size; (void)n_in; (void)out_size;

    // 1. Projection (WMMA)
    gemm_h_kernel<<<(n + 63) / 64, 128, 0, stream>>>(x, W, hbuf, n);
    // 2. Attention halves
    att_kernel<<<(n + 7) / 8, 256, 0, stream>>>(hbuf, att_src, att_dst, a_s, a_d, n);
    // 3. Zero output accumulator
    zero_kernel<<<(n * 32 + 255) / 256, 256, 0, stream>>>((float4*)out, n * 32);
    // 4. Segment-max seed (self-loops)
    node_max_init_kernel<<<(n + 255) / 256, 256, 0, stream>>>(a_s, a_d, mbuf, n);
    // 5. Edge segment max
    edge_max_kernel<<<(e + 255) / 256, 256, 0, stream>>>(src, dst, a_s, a_d, mbuf, e);
    // 6. Denom seed (self-loop p)
    node_denom_init_kernel<<<(n + 255) / 256, 256, 0, stream>>>(a_s, a_d, mbuf, denom, n);
    // 7. Edge exp + denom accumulation
    edge_exp_kernel<<<(e + 255) / 256, 256, 0, stream>>>(src, dst, a_s, a_d, mbuf,
                                                         denom, pbuf, e);
    // 8. Weighted scatter-add (wave per edge)
    edge_agg_kernel<<<(e + 7) / 8, 256, 0, stream>>>(src, dst, hbuf, pbuf, denom, out, e);
    // 9. Self-loop contribution + bias + LayerNorm + PReLU
    final_kernel<<<(n + 7) / 8, 256, 0, stream>>>(hbuf, a_s, a_d, mbuf, denom, bias,
                                                  ln_w, ln_b, prelu_w, out, n);
}
